// TopologyChannel_42992622633779
// MI455X (gfx1250) — compile-verified, matrix operand order checked by reading, not run on hardware
//
#include <hip/hip_runtime.h>
#include <hip/hip_bf16.h>

#define NODES   50000
#define DIMN    128
#define NEDGE0  512000
#define NETOT   562000
#define NTEAMS  5000
#define NLAY    3
#define BNEPS   1e-5f
#define SLOPE   0.2f

typedef __attribute__((ext_vector_type(16))) __bf16 v16bf;
typedef __attribute__((ext_vector_type(8)))  __bf16 v8bf;
typedef __attribute__((ext_vector_type(8)))  float  v8f;

__device__ __forceinline__ __bf16 f2bf(float f) {
    unsigned u = __float_as_uint(f);
    unsigned r = u + 0x7FFFu + ((u >> 16) & 1u);   // round-to-nearest-even
    unsigned short h = (unsigned short)(r >> 16);
    return __builtin_bit_cast(__bf16, h);
}

__device__ __forceinline__ unsigned encF(float f) {
    unsigned u = __float_as_uint(f);
    return (u & 0x80000000u) ? ~u : (u | 0x80000000u);
}
__device__ __forceinline__ float decF(unsigned u) {
    return __uint_as_float((u & 0x80000000u) ? (u & 0x7FFFFFFFu) : ~u);
}

// ---------------- utility ----------------
__global__ void k_fill0(unsigned* __restrict__ p, size_t n) {
    size_t i = (size_t)blockIdx.x * blockDim.x + threadIdx.x;
    if (i < n) p[i] = 0u;
}

__global__ void k_cvt_bf16(const float* __restrict__ src, __bf16* __restrict__ dst, size_t n) {
    size_t i = (size_t)blockIdx.x * blockDim.x + threadIdx.x;
    if (i < n) dst[i] = f2bf(src[i]);
}

// WT[m][c][r] = W[m][r][c]   (bf16, transposed so B fragments load contiguously)
__global__ void k_cvt_weightT(const float* __restrict__ W, __bf16* __restrict__ WT, int nmats) {
    size_t i = (size_t)blockIdx.x * blockDim.x + threadIdx.x;
    size_t tot = (size_t)nmats * DIMN * DIMN;
    if (i >= tot) return;
    int m  = (int)(i / (DIMN * DIMN));
    int rc = (int)(i % (DIMN * DIMN));
    int r = rc / DIMN, c = rc % DIMN;
    WT[(size_t)m * DIMN * DIMN + (size_t)c * DIMN + r] = f2bf(W[i]);
}

// ---------------- bf16 WMMA GEMM: C[M x 128] = A[M x 128] * B + bias ----------------
// BT is B transposed: BT[n][k].  One wave -> one 16x16 tile; block = 8 waves covering all 128 cols.
__global__ __launch_bounds__(256) void k_gemm_bf16(
    const __bf16* __restrict__ A, const __bf16* __restrict__ BT,
    const float* __restrict__ bias, float* __restrict__ C)
{
    const int lane  = threadIdx.x & 31;
    const int wave  = threadIdx.x >> 5;
    const int mbase = blockIdx.x << 4;
    const int nbase = wave << 4;
    const int mr    = lane & 15;
    const int khalf = (lane >> 4) << 3;     // 0 or 8 (16-bit A/B fragment layout)

    const __bf16* arow = A  + (size_t)(mbase + mr) * DIMN;
    const __bf16* brow = BT + (size_t)(nbase + mr) * DIMN;

    v8f acc = {0.f, 0.f, 0.f, 0.f, 0.f, 0.f, 0.f, 0.f};
    #pragma unroll
    for (int k = 0; k < DIMN; k += 32) {
        union Frag { v16bf v; v8bf h[2]; } a, b;
        a.h[0] = *(const v8bf*)(arow + k + khalf);          // K = k+khalf .. +7
        a.h[1] = *(const v8bf*)(arow + k + khalf + 16);     // K = k+khalf+16 .. +23
        b.h[0] = *(const v8bf*)(brow + k + khalf);
        b.h[1] = *(const v8bf*)(brow + k + khalf + 16);
        acc = __builtin_amdgcn_wmma_f32_16x16x32_bf16(
                  false, a.v, false, b.v, (short)0, acc, false, false);
    }

    const int col   = nbase + mr;
    const float bad = bias ? bias[col] : 0.0f;
    const int rbase = mbase + ((lane >> 4) << 3);           // C/D layout: lanes>=16 hold M+8
    #pragma unroll
    for (int v = 0; v < 8; ++v)
        C[(size_t)(rbase + v) * DIMN + col] = acc[v] + bad;
}

// ---------------- edge kernels (wave per edge) ----------------
__device__ __forceinline__ void edge_nodes(const int* __restrict__ ei, int e, int& s, int& t) {
    if (e < NEDGE0) { s = ei[e]; t = ei[NEDGE0 + e]; }
    else            { s = e - NEDGE0; t = s; }
}

__global__ __launch_bounds__(256) void k_edge_score(
    const float* __restrict__ XL, const float* __restrict__ XR,
    const float* __restrict__ att, const int* __restrict__ ei,
    float* __restrict__ esc, unsigned* __restrict__ menc,
    float* __restrict__ outSrc, float* __restrict__ outDst)
{
    int e = blockIdx.x * 8 + (threadIdx.x >> 5);
    if (e >= NETOT) return;
    int lane = threadIdx.x & 31;
    int s, t; edge_nodes(ei, e, s, t);
    float4 l4 = ((const float4*)(XL + (size_t)s * DIMN))[lane];
    float4 r4 = ((const float4*)(XR + (size_t)t * DIMN))[lane];
    float4 a4 = ((const float4*)att)[lane];
    float v0 = l4.x + r4.x; v0 = v0 > 0.f ? v0 : v0 * SLOPE;
    float v1 = l4.y + r4.y; v1 = v1 > 0.f ? v1 : v1 * SLOPE;
    float v2 = l4.z + r4.z; v2 = v2 > 0.f ? v2 : v2 * SLOPE;
    float v3 = l4.w + r4.w; v3 = v3 > 0.f ? v3 : v3 * SLOPE;
    float acc = v0 * a4.x + v1 * a4.y + v2 * a4.z + v3 * a4.w;
    #pragma unroll
    for (int off = 16; off > 0; off >>= 1) acc += __shfl_xor(acc, off, 32);
    if (lane == 0) {
        esc[e] = acc;
        atomicMax(menc + t, encF(acc));
        outSrc[e] = (float)s;
        outDst[e] = (float)t;
    }
}

__global__ void k_softmax_w(const int* __restrict__ ei, float* __restrict__ esc,
                            const unsigned* __restrict__ menc, float* __restrict__ den)
{
    int e = blockIdx.x * blockDim.x + threadIdx.x;
    if (e >= NETOT) return;
    int t = (e < NEDGE0) ? ei[NEDGE0 + e] : (e - NEDGE0);
    float w = expf(esc[e] - decF(menc[t]));
    esc[e] = w;
    atomicAdd(&den[t], w);
}

__global__ void k_alpha(const int* __restrict__ ei, const float* __restrict__ wbuf,
                        const float* __restrict__ den, float* __restrict__ alpha,
                        float* __restrict__ outAlpha)
{
    int e = blockIdx.x * blockDim.x + threadIdx.x;
    if (e >= NETOT) return;
    int t = (e < NEDGE0) ? ei[NEDGE0 + e] : (e - NEDGE0);
    float a = wbuf[e] / den[t];
    alpha[e]    = a;
    outAlpha[e] = a;
}

// agg[dst] += feat[src] * alpha[e]   (wave per edge, float4 per lane)
__global__ __launch_bounds__(256) void k_scatter(
    const float* __restrict__ feat, const float* __restrict__ alpha,
    const int* __restrict__ ei, float* __restrict__ agg)
{
    int e = blockIdx.x * 8 + (threadIdx.x >> 5);
    if (e >= NETOT) return;
    int lane = threadIdx.x & 31;
    int s, t; edge_nodes(ei, e, s, t);
    float a = alpha[e];
    float4 v = ((const float4*)(feat + (size_t)s * DIMN))[lane];
    float* q = agg + (size_t)t * DIMN + lane * 4;
    atomicAdd(q + 0, v.x * a);
    atomicAdd(q + 1, v.y * a);
    atomicAdd(q + 2, v.z * a);
    atomicAdd(q + 3, v.w * a);
}

// ---------------- elementwise ----------------
__global__ void k_add_bias(float* __restrict__ H, const float* __restrict__ b, size_t n) {
    size_t i = (size_t)blockIdx.x * blockDim.x + threadIdx.x;
    if (i < n) H[i] += b[i & (DIMN - 1)];
}

__global__ void k_combine_bf16(const float* __restrict__ H, const float* __restrict__ AGG,
                               const float* __restrict__ eps, int layer,
                               __bf16* __restrict__ xbf, size_t n)
{
    size_t i = (size_t)blockIdx.x * blockDim.x + threadIdx.x;
    if (i >= n) return;
    float sc = 1.0f + eps[layer];
    xbf[i] = f2bf(sc * H[i] + AGG[i]);
}

// ---------------- batch norm ----------------
__global__ __launch_bounds__(256) void k_bn_stats(const float* __restrict__ Z,
                                                  float* __restrict__ sums, float* __restrict__ sqs)
{
    __shared__ float ssum[DIMN];
    __shared__ float ssq[DIMN];
    for (int i = threadIdx.x; i < DIMN; i += blockDim.x) { ssum[i] = 0.f; ssq[i] = 0.f; }
    __syncthreads();
    size_t tot = (size_t)NODES * DIMN;
    size_t stride = (size_t)gridDim.x * blockDim.x;
    for (size_t i = (size_t)blockIdx.x * blockDim.x + threadIdx.x; i < tot; i += stride) {
        float v = Z[i];
        int c = (int)(i & (DIMN - 1));
        atomicAdd(&ssum[c], v);
        atomicAdd(&ssq[c], v * v);
    }
    __syncthreads();
    for (int i = threadIdx.x; i < DIMN; i += blockDim.x) {
        atomicAdd(&sums[i], ssum[i]);
        atomicAdd(&sqs[i],  ssq[i]);
    }
}

__global__ void k_bn_finalize(const float* __restrict__ sums, const float* __restrict__ sqs,
                              const float* __restrict__ g, const float* __restrict__ be,
                              float* __restrict__ scale, float* __restrict__ shift)
{
    int c = threadIdx.x;
    float inv = 1.0f / (float)NODES;
    float mu  = sums[c] * inv;
    float var = sqs[c] * inv - mu * mu;
    float sc  = g[c] * rsqrtf(var + BNEPS);
    scale[c] = sc;
    shift[c] = be[c] - mu * sc;
}

__global__ void k_bn_relu_bf16(const float* __restrict__ Z, const float* __restrict__ scale,
                               const float* __restrict__ shift, __bf16* __restrict__ xbf, size_t n)
{
    size_t i = (size_t)blockIdx.x * blockDim.x + threadIdx.x;
    if (i >= n) return;
    int c = (int)(i & (DIMN - 1));
    float v = fmaxf(Z[i] * scale[c] + shift[c], 0.0f);
    xbf[i] = f2bf(v);
}

__global__ void k_bn_relu_f32(float* __restrict__ H, const float* __restrict__ scale,
                              const float* __restrict__ shift, size_t n)
{
    size_t i = (size_t)blockIdx.x * blockDim.x + threadIdx.x;
    if (i >= n) return;
    int c = (int)(i & (DIMN - 1));
    H[i] = fmaxf(H[i] * scale[c] + shift[c], 0.0f);
}

// ---------------- pooling ----------------
__global__ __launch_bounds__(256) void k_pool(const float* __restrict__ H,
                                              const int* __restrict__ batch,
                                              float* __restrict__ pooled, float* __restrict__ cnt)
{
    int n = blockIdx.x * 8 + (threadIdx.x >> 5);
    if (n >= NODES) return;
    int lane = threadIdx.x & 31;
    int t = batch[n];
    float4 v = ((const float4*)(H + (size_t)n * DIMN))[lane];
    float* q = pooled + (size_t)t * DIMN + lane * 4;
    atomicAdd(q + 0, v.x);
    atomicAdd(q + 1, v.y);
    atomicAdd(q + 2, v.z);
    atomicAdd(q + 3, v.w);
    if (lane == 0) atomicAdd(&cnt[t], 1.0f);
}

__global__ void k_pool_div(float* __restrict__ pooled, const float* __restrict__ cnt, size_t n) {
    size_t i = (size_t)blockIdx.x * blockDim.x + threadIdx.x;
    if (i >= n) return;
    int t = (int)(i >> 7);   // /128
    pooled[i] /= fmaxf(cnt[t], 1.0f);
}

// ---------------- host ----------------
static inline int gridFor(size_t n) { return (int)((n + 255) / 256); }

extern "C" void kernel_launch(void* const* d_in, const int* in_sizes, int n_in,
                              void* d_out, int out_size, void* d_ws, size_t ws_size,
                              hipStream_t stream)
{
    (void)in_sizes; (void)n_in; (void)out_size; (void)ws_size;
    const float* x        = (const float*)d_in[0];
    const int*   ei       = (const int*)  d_in[1];
    const int*   batch    = (const int*)  d_in[2];
    const float* gat_Wl   = (const float*)d_in[3];
    const float* gat_Wr   = (const float*)d_in[4];
    const float* gat_att  = (const float*)d_in[5];
    const float* gat_bias = (const float*)d_in[6];
    const float* gin_W1   = (const float*)d_in[7];
    const float* gin_b1   = (const float*)d_in[8];
    const float* gin_g1   = (const float*)d_in[9];
    const float* gin_be1  = (const float*)d_in[10];
    const float* gin_W2   = (const float*)d_in[11];
    const float* gin_b2   = (const float*)d_in[12];
    const float* gin_g2   = (const float*)d_in[13];
    const float* gin_be2  = (const float*)d_in[14];
    const float* gin_eps  = (const float*)d_in[15];

    const size_t ND = (size_t)NODES * DIMN;

    char*  ws  = (char*)d_ws;
    size_t off = 0;
    auto carve = [&](size_t bytes) -> char* {
        char* p = ws + off;
        off += (bytes + 255) & ~(size_t)255;
        return p;
    };
    __bf16*   XBF  = (__bf16*)  carve(ND * 2);                 // bf16 activations for GEMM A
    __bf16*   WT   = (__bf16*)  carve(8 * 16384 * 2);          // Wl,Wr,W1[0..2],W2[0..2] (transposed)
    float*    XL   = (float*)   carve(ND * 4);
    float*    AGG  = (float*)   carve(ND * 4);                 // XR first, then GIN agg
    float*    H    = (float*)   carve(ND * 4);
    float*    Z    = (float*)   carve(ND * 4);
    float*    ESC  = (float*)   carve((size_t)NETOT * 4);      // scores, then exp-weights
    float*    ALP  = (float*)   carve((size_t)NETOT * 4);
    unsigned* MENC = (unsigned*)carve((size_t)NODES * 4);
    float*    DEN  = (float*)   carve((size_t)NODES * 4);
    float*    BNS  = (float*)   carve(DIMN * 4);               // BNS,BNQ contiguous
    float*    BNQ  = (float*)   carve(DIMN * 4);
    float*    BSC  = (float*)   carve(DIMN * 4);
    float*    BSH  = (float*)   carve(DIMN * 4);
    float*    CNT  = (float*)   carve((size_t)NTEAMS * 4);

    float* out       = (float*)d_out;
    float* outPooled = out;
    float* outSrc    = out + (size_t)NTEAMS * DIMN;
    float* outDst    = outSrc + NETOT;
    float* outAlpha  = outDst + NETOT;

    const int eGrid = NETOT / 8;          // 70250, wave per edge
    const int nGrid = NODES / 8;          // 6250,  wave per node
    const int ewGrid = gridFor(ND);       // elementwise over N*D
    const int mTiles = NODES / 16;        // 3125 GEMM blocks

    // ---- prep: bf16 transposed weights + bf16 x ----
    k_cvt_weightT<<<gridFor(16384),     256, 0, stream>>>(gat_Wl, WT + 0 * 16384, 1);
    k_cvt_weightT<<<gridFor(16384),     256, 0, stream>>>(gat_Wr, WT + 1 * 16384, 1);
    k_cvt_weightT<<<gridFor(3 * 16384), 256, 0, stream>>>(gin_W1, WT + 2 * 16384, 3);
    k_cvt_weightT<<<gridFor(3 * 16384), 256, 0, stream>>>(gin_W2, WT + 5 * 16384, 3);
    k_cvt_bf16<<<ewGrid, 256, 0, stream>>>(x, XBF, ND);

    // ---- GATv2 transforms: xl = x@Wl, xr = x@Wr ----
    k_gemm_bf16<<<mTiles, 256, 0, stream>>>(XBF, WT + 0 * 16384, nullptr, XL);
    k_gemm_bf16<<<mTiles, 256, 0, stream>>>(XBF, WT + 1 * 16384, nullptr, AGG);   // AGG = xr

    // ---- edge scores + segment softmax ----
    k_fill0<<<gridFor(NODES), 256, 0, stream>>>(MENC, NODES);
    k_fill0<<<gridFor(NODES), 256, 0, stream>>>((unsigned*)DEN, NODES);
    k_edge_score<<<eGrid, 256, 0, stream>>>(XL, AGG, gat_att, ei, ESC, MENC, outSrc, outDst);
    k_softmax_w<<<gridFor(NETOT), 256, 0, stream>>>(ei, ESC, MENC, DEN);
    k_alpha<<<gridFor(NETOT), 256, 0, stream>>>(ei, ESC, DEN, ALP, outAlpha);

    // ---- h = segment_sum(xl[src]*alpha) + bias ----
    k_fill0<<<ewGrid, 256, 0, stream>>>((unsigned*)H, ND);
    k_scatter<<<eGrid, 256, 0, stream>>>(XL, ALP, ei, H);
    k_add_bias<<<ewGrid, 256, 0, stream>>>(H, gat_bias, ND);

    // ---- GIN layers ----
    for (int i = 0; i < NLAY; ++i) {
        k_fill0<<<ewGrid, 256, 0, stream>>>((unsigned*)AGG, ND);
        k_scatter<<<eGrid, 256, 0, stream>>>(H, ALP, ei, AGG);
        k_combine_bf16<<<ewGrid, 256, 0, stream>>>(H, AGG, gin_eps, i, XBF, ND);

        // z1 = BNrelu( (1+eps)h+agg @ W1 + b1 )   -> bf16 into XBF
        k_gemm_bf16<<<mTiles, 256, 0, stream>>>(XBF, WT + (2 + i) * 16384, gin_b1 + i * DIMN, Z);
        k_fill0<<<1, 256, 0, stream>>>((unsigned*)BNS, 2 * DIMN);     // BNS+BNQ contiguous
        k_bn_stats<<<512, 256, 0, stream>>>(Z, BNS, BNQ);
        k_bn_finalize<<<1, DIMN, 0, stream>>>(BNS, BNQ, gin_g1 + i * DIMN, gin_be1 + i * DIMN, BSC, BSH);
        k_bn_relu_bf16<<<ewGrid, 256, 0, stream>>>(Z, BSC, BSH, XBF, ND);

        // h = BNrelu( z1 @ W2 + b2 )  (f32, in place in H)
        k_gemm_bf16<<<mTiles, 256, 0, stream>>>(XBF, WT + (5 + i) * 16384, gin_b2 + i * DIMN, H);
        k_fill0<<<1, 256, 0, stream>>>((unsigned*)BNS, 2 * DIMN);
        k_bn_stats<<<512, 256, 0, stream>>>(H, BNS, BNQ);
        k_bn_finalize<<<1, DIMN, 0, stream>>>(BNS, BNQ, gin_g2 + i * DIMN, gin_be2 + i * DIMN, BSC, BSH);
        k_bn_relu_f32<<<ewGrid, 256, 0, stream>>>(H, BSC, BSH, ND);
    }

    // ---- team mean pooling ----
    k_fill0<<<gridFor((size_t)NTEAMS * DIMN), 256, 0, stream>>>((unsigned*)outPooled, (size_t)NTEAMS * DIMN);
    k_fill0<<<gridFor(NTEAMS), 256, 0, stream>>>((unsigned*)CNT, NTEAMS);
    k_pool<<<nGrid, 256, 0, stream>>>(H, batch, outPooled, CNT);
    k_pool_div<<<gridFor((size_t)NTEAMS * DIMN), 256, 0, stream>>>(outPooled, CNT, (size_t)NTEAMS * DIMN);
}
